// AlignedEmbedding_1082331759297
// MI455X (gfx1250) — matrix operand order repaired and checked by reading, not compiled, not run on hardware
//
#include <hip/hip_runtime.h>

// ---------------------------------------------------------------------------
// AlignedEmbedding for MI455X (gfx1250, wave32, WMMA)
//   x_embed = relu(x @ W^T + b); y_embed = relu(y @ W^T + b)
//   S = x_embed @ y_embed^T  (masked -> -inf)
//   alpha = softmax(S); out = alpha @ y
// bf16 WMMA (v_wmma_f32_16x16x32_bf16), f32 accumulation.
// Block tile 128x64, K-step 64, 8 waves; wave w = one 16x64 output strip.
// Double-buffered LDS; pure-bf16 staging via global_load_async_to_lds_b128.
// ---------------------------------------------------------------------------

#define USE_ASYNC_LDS 1

typedef __bf16 v16bf __attribute__((ext_vector_type(16)));
typedef float  v8f   __attribute__((ext_vector_type(8)));

#define LDST 72            // LDS row stride in bf16 elems: 144B rows (16B-aligned, conflict-free)
#define ASZ  (128 * LDST)  // one A buffer (elems)
#define BSZ  (64 * LDST)   // one B buffer (elems)
#define NEG_INF (-__builtin_inff())

// ---- async global->LDS b128 copy (GVS addressing) --------------------------
__device__ __forceinline__ void async_copy_b128(const void* gbase, unsigned voff_bytes,
                                                void* lds_ptr) {
#if USE_ASYNC_LDS
  const unsigned lds_addr = (unsigned)(size_t)lds_ptr;  // flat LDS aperture: [31:0] = LDS offset
  asm volatile("global_load_async_to_lds_b128 %0, %1, %2 offset:0"
               :: "v"(lds_addr), "v"(voff_bytes), "s"(gbase)
               : "memory");
#else
  *(uint4*)lds_ptr = *(const uint4*)((const char*)gbase + voff_bytes);
#endif
}
__device__ __forceinline__ void async_wait_all() {
#if USE_ASYNC_LDS
  asm volatile("s_wait_asynccnt 0x0" ::: "memory");
#endif
}

// ---- WMMA fragment loaders (LDS, bf16) ------------------------------------
// A operand 16x32 (MxK) from row-major tile As[m][k] (stride st).
// Lane L<16: M=L, K in {0..7, 16..23}; lane>=16: M=L-16, K in {8..15, 24..31}.
__device__ __forceinline__ v16bf load_a_frag(const __bf16* As, int mrow, int lane, int st) {
  const int half = lane >> 4, m = lane & 15;
  const __bf16* p = As + (mrow + m) * st + 8 * half;
  union { uint4 u[2]; v16bf v; } r;
  r.u[0] = *(const uint4*)(p);
  r.u[1] = *(const uint4*)(p + 16);
  return r.v;
}
// B operand 32x16 (KxN) staged as Bs[n][k] (stride st).
// Lane L<16: N=L, K in {0..7, 16..23}; lane>=16: N=L-16, K in {8..15, 24..31}.
__device__ __forceinline__ v16bf load_b_frag(const __bf16* Bs, int ncol, int lane, int st) {
  const int n = lane & 15, kb = (lane >> 4) * 8;
  const __bf16* p = Bs + (ncol + n) * st + kb;
  union { uint4 u[2]; v16bf v; } r;
  r.u[0] = *(const uint4*)(p);
  r.u[1] = *(const uint4*)(p + 16);
  return r.v;
}

#define WMMA_BF16(a, b, c) \
  __builtin_amdgcn_wmma_f32_16x16x32_bf16(false, (a), false, (b), (short)0, (c), false, false)

// Inner compute step on one staged tile pair: 2 K-substeps x 4 N-tiles.
__device__ __forceinline__ void mma_tile(const __bf16* As, const __bf16* Bs,
                                         int msub, int lane, v8f acc[4]) {
#pragma unroll
  for (int kk = 0; kk < 64; kk += 32) {
    const v16bf a = load_a_frag(As + kk, msub, lane, LDST);
    v16bf bb[4];
#pragma unroll
    for (int j = 0; j < 4; ++j) bb[j] = load_b_frag(Bs + kk, j * 16, lane, LDST);
#pragma unroll
    for (int j = 0; j < 4; ++j) acc[j] = WMMA_BF16(a, bb[j], acc[j]);
  }
}

// ---- K1: W f32 -> bf16 -----------------------------------------------------
__global__ __launch_bounds__(256) void cvt_w_kernel(const float* __restrict__ W,
                                                    __bf16* __restrict__ Wb) {
  const size_t i = (size_t)blockIdx.x * 1024 + threadIdx.x * 4;
  const float4 f = *(const float4*)(W + i);
  __bf16* d = Wb + i;
  d[0] = (__bf16)f.x; d[1] = (__bf16)f.y; d[2] = (__bf16)f.z; d[3] = (__bf16)f.w;
}

// ---- K2: Out = relu(A @ W^T + b); A f32 [M,1024], W bf16 [1024,1024] -------
__global__ __launch_bounds__(256) void proj_gemm_relu_kernel(
    const float* __restrict__ A, const __bf16* __restrict__ Wb,
    const float* __restrict__ bias, __bf16* __restrict__ Out) {
  __shared__ __align__(16) __bf16 As[2 * ASZ];
  __shared__ __align__(16) __bf16 Bs[2 * BSZ];
  const int tid  = threadIdx.x;
  const int lane = tid & 31, wave = tid >> 5;
  const int m0 = blockIdx.y * 128, n0 = blockIdx.x * 64;
  const int msub = wave * 16;

  v8f acc[4];
#pragma unroll
  for (int j = 0; j < 4; ++j) acc[j] = (v8f){0.f, 0.f, 0.f, 0.f, 0.f, 0.f, 0.f, 0.f};

  auto stage = [&](int buf, int k0) {
    __bf16* Ad = As + buf * ASZ;
    __bf16* Bd = Bs + buf * BSZ;
    // B (W rows n0..n0+63, 64 K): 512 x b128 async direct to LDS
    int idx = tid;
#pragma unroll
    for (int it = 0; it < 2; ++it, idx += 256) {
      const int r = idx >> 3, c8 = (idx & 7) * 8;
      async_copy_b128(Wb, (unsigned)(2u * ((unsigned)(n0 + r) * 1024u + k0 + c8)),
                      Bd + r * LDST + c8);
    }
    // A (128 x 64 f32 -> bf16): 8 float4 per thread
    idx = tid;
#pragma unroll
    for (int it = 0; it < 8; ++it, idx += 256) {
      const int r = idx >> 4, c4 = (idx & 15) * 4;
      const float4 f = *(const float4*)(A + (size_t)(m0 + r) * 1024 + k0 + c4);
      __bf16* d = Ad + r * LDST + c4;
      d[0] = (__bf16)f.x; d[1] = (__bf16)f.y; d[2] = (__bf16)f.z; d[3] = (__bf16)f.w;
    }
  };

  stage(0, 0);
  async_wait_all();
  __syncthreads();
  for (int k0 = 0; k0 < 1024; k0 += 64) {
    const int cur = (k0 >> 6) & 1;
    if (k0 + 64 < 1024) stage(cur ^ 1, k0 + 64);        // overlap next tile with compute
    mma_tile(As + cur * ASZ, Bs + cur * BSZ, msub, lane, acc);
    async_wait_all();
    __syncthreads();
  }

  // epilogue: +bias, relu, bf16. C layout: VGPR r -> row r + 8*(lane>>4), col lane&15.
  const int half = lane >> 4, n = lane & 15;
#pragma unroll
  for (int j = 0; j < 4; ++j) {
    const float bv = bias[n0 + j * 16 + n];
#pragma unroll
    for (int r = 0; r < 8; ++r) {
      const int row = m0 + msub + r + 8 * half;
      float v = acc[j][r] + bv;
      v = v > 0.f ? v : 0.f;
      Out[(size_t)row * 1024 + n0 + j * 16 + n] = (__bf16)v;
    }
  }
}

// ---- K3: S[b] = Xe[b] @ Ye[b]^T, masked; f32 out ---------------------------
__global__ __launch_bounds__(256) void scores_gemm_mask_kernel(
    const __bf16* __restrict__ Xe, const __bf16* __restrict__ Ye,
    const unsigned char* __restrict__ mask, float* __restrict__ S) {
  const int b = blockIdx.z;
  const __bf16* Ab = Xe + (size_t)b * 1024 * 1024;
  const __bf16* Bb = Ye + (size_t)b * 1024 * 1024;
  float* Sb = S + (size_t)b * 1024 * 1024;
  const unsigned char* mb = mask + (size_t)b * 1024;

  __shared__ __align__(16) __bf16 As[2 * ASZ];
  __shared__ __align__(16) __bf16 Bs[2 * BSZ];
  const int tid  = threadIdx.x;
  const int lane = tid & 31, wave = tid >> 5;
  const int m0 = blockIdx.y * 128, n0 = blockIdx.x * 64;
  const int msub = wave * 16;

  v8f acc[4];
#pragma unroll
  for (int j = 0; j < 4; ++j) acc[j] = (v8f){0.f, 0.f, 0.f, 0.f, 0.f, 0.f, 0.f, 0.f};

  auto stage = [&](int buf, int k0) {
    __bf16* Ad = As + buf * ASZ;
    __bf16* Bd = Bs + buf * BSZ;
    int idx = tid;  // A: 128x64 bf16 = 1024 x b128 (4/thread)
#pragma unroll
    for (int it = 0; it < 4; ++it, idx += 256) {
      const int r = idx >> 3, c8 = (idx & 7) * 8;
      async_copy_b128(Ab, (unsigned)(2u * ((unsigned)(m0 + r) * 1024u + k0 + c8)),
                      Ad + r * LDST + c8);
    }
    idx = tid;      // B: 64x64 bf16 = 512 x b128 (2/thread)
#pragma unroll
    for (int it = 0; it < 2; ++it, idx += 256) {
      const int r = idx >> 3, c8 = (idx & 7) * 8;
      async_copy_b128(Bb, (unsigned)(2u * ((unsigned)(n0 + r) * 1024u + k0 + c8)),
                      Bd + r * LDST + c8);
    }
  };

  stage(0, 0);
  async_wait_all();
  __syncthreads();
  for (int k0 = 0; k0 < 1024; k0 += 64) {
    const int cur = (k0 >> 6) & 1;
    if (k0 + 64 < 1024) stage(cur ^ 1, k0 + 64);
    mma_tile(As + cur * ASZ, Bs + cur * BSZ, msub, lane, acc);
    async_wait_all();
    __syncthreads();
  }

  const int half = lane >> 4, n = lane & 15;
#pragma unroll
  for (int j = 0; j < 4; ++j) {
    const int col = n0 + j * 16 + n;
    const bool msk = mb[col] != 0;
#pragma unroll
    for (int r = 0; r < 8; ++r) {
      const int row = m0 + msub + r + 8 * half;
      Sb[(size_t)row * 1024 + col] = msk ? NEG_INF : acc[j][r];
    }
  }
}

// ---- K4: row softmax; writes alpha as bf16 in-place at row start ----------
__global__ __launch_bounds__(256) void softmax_rows_kernel(float* __restrict__ S) {
  float* p = S + (size_t)blockIdx.x * 1024;
  const int tid = threadIdx.x, lane = tid & 31, wave = tid >> 5;
  __shared__ float red[8];

  float v[4];
  float mx = NEG_INF;
#pragma unroll
  for (int j = 0; j < 4; ++j) { v[j] = p[tid + 256 * j]; mx = fmaxf(mx, v[j]); }
#pragma unroll
  for (int o = 16; o > 0; o >>= 1) mx = fmaxf(mx, __shfl_xor(mx, o, 32));
  if (lane == 0) red[wave] = mx;
  __syncthreads();
  if (tid == 0) {
    float m = red[0];
#pragma unroll
    for (int w = 1; w < 8; ++w) m = fmaxf(m, red[w]);
    red[0] = m;
  }
  __syncthreads();
  mx = red[0];
  __syncthreads();

  float s = 0.f;
#pragma unroll
  for (int j = 0; j < 4; ++j) { v[j] = __expf(v[j] - mx); s += v[j]; }  // exp(-inf)=0
#pragma unroll
  for (int o = 16; o > 0; o >>= 1) s += __shfl_xor(s, o, 32);
  if (lane == 0) red[wave] = s;
  __syncthreads();
  if (tid == 0) {
    float t = red[0];
#pragma unroll
    for (int w = 1; w < 8; ++w) t += red[w];
    red[0] = t;
  }
  __syncthreads();
  const float inv = 1.0f / red[0];
  __syncthreads();  // all f32 row reads complete before aliasing bf16 writes

  __bf16* a = (__bf16*)p;  // alpha occupies first half of each f32 row
#pragma unroll
  for (int j = 0; j < 4; ++j) a[tid + 256 * j] = (__bf16)(v[j] * inv);
}

// ---- K5: out[b] = alpha[b] @ y[b]; alpha bf16 (row stride 2048), y f32 -----
__global__ __launch_bounds__(256) void out_gemm_kernel(
    const float* __restrict__ S /* alpha embedded */, const float* __restrict__ Y,
    float* __restrict__ Out) {
  const int b = blockIdx.z;
  const __bf16* Aa = (const __bf16*)(S + (size_t)b * 1024 * 1024);  // ld = 2048 bf16
  const float* Yb = Y + (size_t)b * 1024 * 1024;
  float* Ob = Out + (size_t)b * 1024 * 1024;

  __shared__ __align__(16) __bf16 As[2 * ASZ];
  __shared__ __align__(16) __bf16 Bs[2 * BSZ];
  const int tid  = threadIdx.x;
  const int lane = tid & 31, wave = tid >> 5;
  const int m0 = blockIdx.y * 128, n0 = blockIdx.x * 64;
  const int msub = wave * 16;

  v8f acc[4];
#pragma unroll
  for (int j = 0; j < 4; ++j) acc[j] = (v8f){0.f, 0.f, 0.f, 0.f, 0.f, 0.f, 0.f, 0.f};

  auto stage = [&](int buf, int k0) {
    __bf16* Ad = As + buf * ASZ;
    __bf16* Bd = Bs + buf * BSZ;
    int idx = tid;  // alpha tile 128x64 bf16 (row stride 2048), async
#pragma unroll
    for (int it = 0; it < 4; ++it, idx += 256) {
      const int r = idx >> 3, c8 = (idx & 7) * 8;
      async_copy_b128(Aa, (unsigned)(2u * ((unsigned)(m0 + r) * 2048u + k0 + c8)),
                      Ad + r * LDST + c8);
    }
    idx = tid;      // y tile 64(t) x 64(e) f32, transposed + converted -> Bs[e][t]
#pragma unroll
    for (int it = 0; it < 2; ++it, idx += 256) {
      const int tr = idx >> 3, ec = (idx & 7) * 8;
      const float4 f0 = *(const float4*)(Yb + (size_t)(k0 + tr) * 1024 + n0 + ec);
      const float4 f1 = *(const float4*)(Yb + (size_t)(k0 + tr) * 1024 + n0 + ec + 4);
      Bd[(ec + 0) * LDST + tr] = (__bf16)f0.x;
      Bd[(ec + 1) * LDST + tr] = (__bf16)f0.y;
      Bd[(ec + 2) * LDST + tr] = (__bf16)f0.z;
      Bd[(ec + 3) * LDST + tr] = (__bf16)f0.w;
      Bd[(ec + 4) * LDST + tr] = (__bf16)f1.x;
      Bd[(ec + 5) * LDST + tr] = (__bf16)f1.y;
      Bd[(ec + 6) * LDST + tr] = (__bf16)f1.z;
      Bd[(ec + 7) * LDST + tr] = (__bf16)f1.w;
    }
  };

  stage(0, 0);
  async_wait_all();
  __syncthreads();
  for (int k0 = 0; k0 < 1024; k0 += 64) {
    const int cur = (k0 >> 6) & 1;
    if (k0 + 64 < 1024) stage(cur ^ 1, k0 + 64);
    mma_tile(As + cur * ASZ, Bs + cur * BSZ, msub, lane, acc);
    async_wait_all();
    __syncthreads();
  }

  const int half = lane >> 4, n = lane & 15;
#pragma unroll
  for (int j = 0; j < 4; ++j) {
#pragma unroll
    for (int r = 0; r < 8; ++r) {
      const int row = m0 + msub + r + 8 * half;
      Ob[(size_t)row * 1024 + n0 + j * 16 + n] = acc[j][r];
    }
  }
}

// ---------------------------------------------------------------------------
extern "C" void kernel_launch(void* const* d_in, const int* in_sizes, int n_in,
                              void* d_out, int out_size, void* d_ws, size_t ws_size,
                              hipStream_t stream) {
  // setup_inputs order: x, y, y_mask, W, b
  const float*         x    = (const float*)d_in[0];         // [16,1024,1024]
  const float*         y    = (const float*)d_in[1];         // [16,1024,1024]
  const unsigned char* mask = (const unsigned char*)d_in[2]; // [16,1024] bool
  const float*         W    = (const float*)d_in[3];         // [1024,1024]
  const float*         bias = (const float*)d_in[4];         // [1024]
  float*               out  = (float*)d_out;                 // [16,1024,1024]

  // workspace layout (130 MB total)
  char* ws = (char*)d_ws;
  __bf16* Wb = (__bf16*)(ws);                                  //  2 MB
  __bf16* Xe = (__bf16*)(ws + (size_t)(2)  * 1024 * 1024);     // 32 MB
  __bf16* Ye = (__bf16*)(ws + (size_t)(34) * 1024 * 1024);     // 32 MB
  float*  S  = (float*) (ws + (size_t)(66) * 1024 * 1024);     // 64 MB (scores -> alpha)

  cvt_w_kernel<<<1024, 256, 0, stream>>>(W, Wb);
  proj_gemm_relu_kernel<<<dim3(16, 128), 256, 0, stream>>>(x, Wb, bias, Xe);
  proj_gemm_relu_kernel<<<dim3(16, 128), 256, 0, stream>>>(y, Wb, bias, Ye);
  scores_gemm_mask_kernel<<<dim3(16, 8, 16), 256, 0, stream>>>(Xe, Ye, mask, S);
  softmax_rows_kernel<<<16 * 1024, 256, 0, stream>>>(S);
  out_gemm_kernel<<<dim3(16, 8, 16), 256, 0, stream>>>(S, y, out);
}
